// Simba_43482248904984
// MI455X (gfx1250) — matrix operand order repaired and checked by reading, not compiled, not run on hardware
//
#include <hip/hip_runtime.h>
#include <hip/hip_bf16.h>
#include <math.h>

#define BB 4
#define NN 8192
#define CC 256
#define HH 512
#define KK 16
#define K2C (2*CC)              // 512 = contraction dim of first GEMM
#define NP 8                    // points per workgroup (== waves per block)
#define HCHUNK 128              // W1 columns resident per LDS buffer
#define NCHUNKS (HH / HCHUNK)   // 4
#define NTILES (HCHUNK / 16)    // 8
#define KSTEPS (K2C / 32)       // 16
#define TPB 256

typedef __attribute__((ext_vector_type(16))) __bf16 v16bf;
typedef __attribute__((ext_vector_type(8)))  __bf16 v8bf;
typedef __attribute__((ext_vector_type(8)))  float  v8f;
typedef __attribute__((ext_vector_type(4)))  unsigned int v4u;
typedef __attribute__((ext_vector_type(4)))  int v4i;
typedef __attribute__((ext_vector_type(8)))  int v8i;

#if __has_builtin(__builtin_amdgcn_tensor_load_to_lds)
#define HAVE_TDM 1
#else
#define HAVE_TDM 0
#endif

// ---------------- WMMA helpers (CDNA5 16x16x32 bf16, f32 accum) ----------------

__device__ __forceinline__ v8f vzero8() {
  v8f z;
#pragma unroll
  for (int i = 0; i < 8; ++i) z[i] = 0.0f;
  return z;
}

// A-matrix 16x32 bf16 fragment from a row-major LDS tile (used for stage-3 Hlds).
// ISA layout: row = lane&15, K = kb..kb+7 (v0-3) and kb+16..kb+23 (v4-7), kb = 8*(lane>=16).
__device__ __forceinline__ v16bf frag_a_lds(const __bf16* tile, int row_stride, int lane) {
  int r  = lane & 15;
  int kb = (lane >> 4) << 3;
  const __bf16* p = tile + r * row_stride + kb;
  v8bf lo = *(const v8bf*)(p);
  v8bf hi = *(const v8bf*)(p + 16);
  return __builtin_shufflevector(lo, hi, 0,1,2,3,4,5,6,7,8,9,10,11,12,13,14,15);
}

// B-matrix 32x16 bf16 fragment: column n = lane&15 holds K = kb..kb+7 and kb+16..kb+23,
// kb = 8*(lane>=16). Weight rows are contiguous over
// the contraction dim, so each half is a contiguous 16-byte load.
__device__ __forceinline__ v16bf frag_b_lds(const __bf16* tile, int col_stride, int lane) {
  int n  = lane & 15;
  int kb = (lane >> 4) << 3;
  const __bf16* p = tile + n * col_stride + kb;
  v8bf lo = *(const v8bf*)(p);
  v8bf hi = *(const v8bf*)(p + 16);
  return __builtin_shufflevector(lo, hi, 0,1,2,3,4,5,6,7,8,9,10,11,12,13,14,15);
}

// B fragment from global bf16 weights (W2bf), same per-lane layout.
__device__ __forceinline__ v16bf frag_b_gbf(const __bf16* __restrict__ W, int cobase,
                                            int k0, int row_stride, int lane) {
  int co = cobase + (lane & 15);
  int kb = k0 + ((lane >> 4) << 3);
  const __bf16* p = W + (size_t)co * row_stride + kb;
  v8bf lo = *(const v8bf*)(p);
  v8bf hi = *(const v8bf*)(p + 16);
  return __builtin_shufflevector(lo, hi, 0,1,2,3,4,5,6,7,8,9,10,11,12,13,14,15);
}

__device__ __forceinline__ v8f wmma_bf16(v16bf a, v16bf b, v8f c) {
  return __builtin_amdgcn_wmma_f32_16x16x32_bf16(false, a, false, b, (short)0, c,
                                                 false, false);
}

__device__ __forceinline__ float gelu_exact(float x) {
  return 0.5f * x * (1.0f + erff(x * 0.70710678118654752f));
}

#if HAVE_TDM
// TDM descriptor: copy `nrows` contiguous 512-elem bf16 rows (row-major, stride 512)
// from global to LDS. D# fields per CDNA5 ISA ch.8: group0 = {count=1, lds_addr,
// global_addr[56:0], type=2}; group1 = {data_size=2B, tensor_dim0/1, tile_dim0/1, stride0}.
__device__ __forceinline__ void tdm_load_rows(const __bf16* gsrc, unsigned lds_off, int nrows) {
  unsigned long long ga = (unsigned long long)gsrc;
  v4u g0 = { 1u,                                    // count=1 (valid), user mode
             lds_off,                               // lds_addr (bytes)
             (unsigned)(ga & 0xFFFFFFFFull),        // global_addr[31:0]
             (unsigned)((ga >> 32) & 0x1FFFFFFull) | 0x80000000u };  // addr[56:32] | type=2
  v8i g1;
  g1[0] = 1 << 16;                 // workgroup_mask=0, data_size=1 (2 bytes)
  g1[1] = (int)(K2C << 16);        // tensor_dim0[15:0] = 512
  g1[2] = (int)(HH  << 16);        // tensor_dim0 hi=0 | tensor_dim1[15:0] = 512
  g1[3] = (int)(K2C << 16);        // tensor_dim1 hi=0 | tile_dim0 = 512 (full row)
  g1[4] = nrows;                   // tile_dim1 = rows, tile_dim2 = 0
  g1[5] = K2C;                     // tensor_dim0_stride[31:0] = 512
  g1[6] = 0;                       // stride0 hi | stride1 lo
  g1[7] = 0;
  v4i z4 = {0, 0, 0, 0};
#if defined(__clang_major__) && (__clang_major__ >= 23)
  v8i z8 = {0, 0, 0, 0, 0, 0, 0, 0};
  __builtin_amdgcn_tensor_load_to_lds(g0, g1, z4, z4, z8, 0);
#else
  __builtin_amdgcn_tensor_load_to_lds(g0, g1, z4, z4, 0);
#endif
}
#endif

// ---------------- Kernel 1: brute-force KNN (k=16) ----------------

__global__ __launch_bounds__(256, 2)
void knn_kernel(const float* __restrict__ XYZ, int* __restrict__ IDX) {
  __shared__ float tx[256], ty[256], tz[256];
  int b = blockIdx.x >> 5;
  int q = ((blockIdx.x & 31) << 8) + threadIdx.x;
  const float* base = XYZ + (size_t)b * NN * 3;
  float qx = base[q*3+0], qy = base[q*3+1], qz = base[q*3+2];

  float bd[KK]; int bi[KK];
#pragma unroll
  for (int t = 0; t < KK; ++t) { bd[t] = 3.0e38f; bi[t] = 0; }

  for (int tile = 0; tile < NN/256; ++tile) {
    int c0 = tile << 8;
    __syncthreads();
    tx[threadIdx.x] = base[(c0+threadIdx.x)*3+0];
    ty[threadIdx.x] = base[(c0+threadIdx.x)*3+1];
    tz[threadIdx.x] = base[(c0+threadIdx.x)*3+2];
    __syncthreads();
    for (int j = 0; j < 256; ++j) {
      float dx = tx[j]-qx, dy = ty[j]-qy, dz = tz[j]-qz;
      float d = dx*dx + dy*dy + dz*dz;
      if (d < bd[KK-1]) {
        bd[KK-1] = d; bi[KK-1] = c0 + j;
#pragma unroll
        for (int t = KK-1; t > 0; --t) {
          if (bd[t] < bd[t-1]) {
            float td = bd[t]; bd[t] = bd[t-1]; bd[t-1] = td;
            int   ti = bi[t]; bi[t] = bi[t-1]; bi[t-1] = ti;
          }
        }
      }
    }
  }
  int* o = IDX + ((size_t)b*NN + q) * KK;
#pragma unroll
  for (int t = 0; t < KK; ++t) o[t] = bi[t];
}

// ---------------- Kernel 2: fp32 -> bf16 weight conversion ----------------

__global__ __launch_bounds__(256)
void cvt_weights(const float* __restrict__ W1, const float* __restrict__ W2,
                 __bf16* __restrict__ W1bf, __bf16* __restrict__ W2bf) {
  int i = blockIdx.x * 256 + threadIdx.x;
  if (i < HH * K2C) W1bf[i] = (__bf16)W1[i];
  if (i < CC * HH)  W2bf[i] = (__bf16)W2[i];
}

// ---------------- Kernel 3: fused gather -> GEMM1 -> gelu -> max_k -> GEMM2 ----------------

__global__ __launch_bounds__(TPB, 1)
void fused_edge_mlp(const float* __restrict__ X,  const __bf16* __restrict__ W1bf,
                    const float* __restrict__ B1, const __bf16* __restrict__ W2bf,
                    const float* __restrict__ B2, const int* __restrict__ IDX,
                    float* __restrict__ OUT) {
  __shared__ __attribute__((aligned(16))) __bf16 Wlds[2 * HCHUNK * K2C]; // 256 KB double buffer
  __shared__ __attribute__((aligned(16))) __bf16 Hlds[16 * HH];          //  16 KB, rows 8..15 = 0

  const int tid  = threadIdx.x;
  const int lane = tid & 31;
  const int wave = tid >> 5;                 // == point index within the block
  const int b    = blockIdx.x / (NN / NP);
  const int n0   = (blockIdx.x % (NN / NP)) * NP;

  for (int e = tid; e < 8*HH; e += TPB)      // zero padding rows of stage-3 A tile
    Hlds[8*HH + e] = (__bf16)0.0f;

  // Stage 1: build all 16 A fragments for this wave's point directly in registers.
  // WMMA A layout: row = lane&15 == neighbor j; lanes l / l+16 cover the two K halves.
  const float* Xb  = X + (size_t)b * NN * CC;
  const int row    = lane & 15;
  const int khalf  = (lane >> 4) << 3;       // 0 or 8
  const int nbr    = IDX[((size_t)b*NN + n0 + wave) * KK + row];
  const float* cen = Xb + (size_t)(n0 + wave) * CC;
  const float* nei = Xb + (size_t)nbr * CC;

  v16bf afrag[KSTEPS];
#pragma unroll
  for (int ks = 0; ks < KSTEPS; ++ks) {
    const int kb = (ks & 7) * 32 + khalf;    // base within the 256-wide half
    float v[16];
#pragma unroll
    for (int g = 0; g < 2; ++g) {
      const int k0 = kb + g * 16;
      float4 c0 = *(const float4*)(cen + k0);
      float4 c1 = *(const float4*)(cen + k0 + 4);
      if (ks < 8) {                          // compile-time uniform after unroll
        float4 m0 = *(const float4*)(nei + k0);
        float4 m1 = *(const float4*)(nei + k0 + 4);
        v[g*8+0] = m0.x - c0.x; v[g*8+1] = m0.y - c0.y;
        v[g*8+2] = m0.z - c0.z; v[g*8+3] = m0.w - c0.w;
        v[g*8+4] = m1.x - c1.x; v[g*8+5] = m1.y - c1.y;
        v[g*8+6] = m1.z - c1.z; v[g*8+7] = m1.w - c1.w;
      } else {
        v[g*8+0] = c0.x; v[g*8+1] = c0.y; v[g*8+2] = c0.z; v[g*8+3] = c0.w;
        v[g*8+4] = c1.x; v[g*8+5] = c1.y; v[g*8+6] = c1.z; v[g*8+7] = c1.w;
      }
    }
    v16bf f;
#pragma unroll
    for (int i = 0; i < 16; ++i) f[i] = (__bf16)v[i];
    afrag[ks] = f;
  }

#if HAVE_TDM
  const unsigned wlds_base = (unsigned)(unsigned long long)&Wlds[0];
  if (tid == 0) tdm_load_rows(W1bf, wlds_base, HCHUNK);   // prefetch chunk 0
#endif

  // Stage 2: h = gelu(g @ W1^T + b1); max over k. W1 streamed via TDM double buffer.
#pragma unroll 1
  for (int ch = 0; ch < NCHUNKS; ++ch) {
    const int hbase = ch * HCHUNK;
    const __bf16* Wbuf = &Wlds[(ch & 1) * HCHUNK * K2C];
    __syncthreads();                         // prior chunk fully consumed
#if HAVE_TDM
    if (tid == 0) {
      if (ch + 1 < NCHUNKS) {
        tdm_load_rows(W1bf + (size_t)(hbase + HCHUNK) * K2C,
                      wlds_base + (unsigned)(((ch + 1) & 1) * HCHUNK * K2C * 2), HCHUNK);
        __builtin_amdgcn_s_wait_tensorcnt((short)1);  // chunk ch done, ch+1 in flight
      } else {
        __builtin_amdgcn_s_wait_tensorcnt((short)0);
      }
    }
#else
    {
      const v8bf* src = (const v8bf*)(W1bf + (size_t)hbase * K2C);
      v8bf* dst = (v8bf*)Wbuf;
      for (int it = tid; it < (HCHUNK * K2C) / 8; it += TPB) dst[it] = src[it];
    }
#endif
    __syncthreads();                         // current buffer ready

    v8f acc[NTILES];
#pragma unroll
    for (int nt = 0; nt < NTILES; ++nt) acc[nt] = vzero8();

#pragma unroll
    for (int ks = 0; ks < KSTEPS; ++ks) {    // fully unrolled: afrag stays in VGPRs
      v16bf a = afrag[ks];
#pragma unroll
      for (int nt = 0; nt < NTILES; ++nt) {
        v16bf bm = frag_b_lds(Wbuf + (size_t)(nt*16)*K2C + ks*32, K2C, lane);
        acc[nt] = wmma_bf16(a, bm, acc[nt]);
      }
    }

    // Epilogue: +b1, exact gelu, max over the 16 k-rows (8 VGPRs + lane^16 partner).
#pragma unroll
    for (int nt = 0; nt < NTILES; ++nt) {
      int col = hbase + nt*16 + (lane & 15);
      float bb = B1[col];
      float m = -3.0e38f;
#pragma unroll
      for (int r = 0; r < 8; ++r)
        m = fmaxf(m, gelu_exact(acc[nt][r] + bb));
      m = fmaxf(m, __shfl_xor(m, 16, 32));
      if (lane < 16) Hlds[wave*HH + col] = (__bf16)m;
    }
  }
  __syncthreads();

  // Stage 3: out = h_max @ W2^T + b2 via WMMA (M=16, rows 0..7 valid). 2 N-tiles/wave.
#pragma unroll
  for (int t = 0; t < 2; ++t) {
    int cobase = (wave*2 + t) * 16;
    v8f acc2 = vzero8();
#pragma unroll
    for (int ks = 0; ks < KSTEPS; ++ks) {
      v16bf a  = frag_a_lds(&Hlds[0] + ks*32, HH, lane);
      v16bf bm = frag_b_gbf(W2bf, cobase, ks*32, HH, lane);
      acc2 = wmma_bf16(a, bm, acc2);
    }
    float bb = B2[cobase + (lane & 15)];
    if (lane < 16) {
#pragma unroll
      for (int r = 0; r < 8; ++r)
        OUT[((size_t)b*NN + n0 + r)*CC + cobase + lane] = acc2[r] + bb;
    }
  }
}

// ---------------- Host entry ----------------

extern "C" void kernel_launch(void* const* d_in, const int* in_sizes, int n_in,
                              void* d_out, int out_size, void* d_ws, size_t ws_size,
                              hipStream_t stream) {
  (void)in_sizes; (void)n_in; (void)out_size; (void)ws_size;
  const float* X   = (const float*)d_in[0];
  const float* XYZ = (const float*)d_in[1];
  const float* W1  = (const float*)d_in[2];
  const float* B1  = (const float*)d_in[3];
  const float* W2  = (const float*)d_in[4];
  const float* B2  = (const float*)d_in[5];

  char* ws = (char*)d_ws;
  int*    IDX  = (int*)ws;                                    // 2 MB
  __bf16* W1bf = (__bf16*)(ws + (size_t)BB*NN*KK*4);          // 512 KB
  __bf16* W2bf = (__bf16*)(ws + (size_t)BB*NN*KK*4 + (size_t)HH*K2C*2);  // 256 KB
  float* OUT = (float*)d_out;

  knn_kernel<<<BB * (NN/256), 256, 0, stream>>>(XYZ, IDX);
  cvt_weights<<<(HH*K2C + 255)/256, 256, 0, stream>>>(W1, W2, W1bf, W2bf);
  fused_edge_mlp<<<BB * (NN/NP), TPB, 0, stream>>>(X, W1bf, B1, W2bf, B2, IDX, OUT);
}
